// MultiHeadAttention_52793738002650
// MI455X (gfx1250) — matrix operand order repaired and checked by reading, not compiled, not run on hardware
//
#include <hip/hip_runtime.h>
#include <hip/hip_bf16.h>
#include <stdint.h>

// ---- problem constants (reference: B=4, N=2048, EMB=512, HEADS=8) ----
#define EMB   512
#define HEADS 8
#define HD    64
#define BATCH 4
#define SEQ   2048
#define ROWS  (BATCH * SEQ)   // 8192

typedef __attribute__((ext_vector_type(16))) __bf16 v16bf;
typedef __attribute__((ext_vector_type(8)))  float  v8f;

union Frag32B { uint4 u[2]; v16bf v; };

__device__ __forceinline__ uint16_t f2bf(float f) {
  union { float f; uint32_t u; } a; a.f = f;
  uint32_t u = a.u;
  return (uint16_t)((u + 0x7FFFu + ((u >> 16) & 1u)) >> 16);   // RNE
}

// A fragment (16x32 bf16, MxK): lane holds row m=lane%16; within-lane K order is
// contiguous [c0..c0+15], c0 = (lane>=16)*16.
__device__ __forceinline__ v16bf loadA(const uint16_t* base, int laneHi) {
  const uint4* p = reinterpret_cast<const uint4*>(base);
  Frag32B f;
  f.u[0] = p[2 * laneHi];       // K = c0 .. c0+7
  f.u[1] = p[2 * laneHi + 1];   // K = c0+8 .. c0+15
  return f.v;
}
// B fragment (32x16 bf16, KxN): lane holds col n=lane%16; 16 contiguous K
// starting at (lane>=16)*16. base -> already offset to that chunk.
__device__ __forceinline__ v16bf loadB(const uint16_t* base) {
  const uint4* p = reinterpret_cast<const uint4*>(base);
  Frag32B f;
  f.u[0] = p[0];
  f.u[1] = p[1];
  return f.v;
}

__device__ __forceinline__ v8f wmma_bf16(v16bf a, v16bf b, v8f c) {
  return __builtin_amdgcn_wmma_f32_16x16x32_bf16(false, a, false, b, (short)0, c, false, false);
}

// CDNA5 async copy global->LDS (16B per lane, ASYNCcnt-tracked).
// lds_off: byte offset within LDS (low 32 bits of generic shared address).
__device__ __forceinline__ void async_load_b128(uint32_t lds_off, const void* gptr) {
  asm volatile("global_load_async_to_lds_b128 %0, %1, off"
               :: "v"(lds_off), "v"((uint64_t)(uintptr_t)gptr)
               : "memory");
}
__device__ __forceinline__ void wait_async0() {
  asm volatile("s_wait_asynccnt 0" ::: "memory");
}
__device__ __forceinline__ uint32_t lds_off_of(const void* p) {
  return (uint32_t)(uintptr_t)p;   // LDS aperture: addr[31:0] = LDS byte offset
}

// reductions over a 16-lane group (C-layout rows live across lanes 0..15 / 16..31)
__device__ __forceinline__ float rmax16(float v) {
  v = fmaxf(v, __shfl_xor(v, 1, 32));
  v = fmaxf(v, __shfl_xor(v, 2, 32));
  v = fmaxf(v, __shfl_xor(v, 4, 32));
  v = fmaxf(v, __shfl_xor(v, 8, 32));
  return v;
}
__device__ __forceinline__ float rsum16(float v) {
  v += __shfl_xor(v, 1, 32);
  v += __shfl_xor(v, 2, 32);
  v += __shfl_xor(v, 4, 32);
  v += __shfl_xor(v, 8, 32);
  return v;
}

// ---------------- prep: fp32 -> bf16 ----------------
__global__ __launch_bounds__(256) void cvt_x_kernel(const float* __restrict__ x,
                                                    uint16_t* __restrict__ xb) {
  int i = blockIdx.x * blockDim.x + threadIdx.x;
  float4 v = reinterpret_cast<const float4*>(x)[i];
  ushort4 o;
  o.x = f2bf(v.x); o.y = f2bf(v.y); o.z = f2bf(v.z); o.w = f2bf(v.w);
  reinterpret_cast<ushort4*>(xb)[i] = o;
}

// transpose + convert weights: Wt[n][k] = W[k][n]
__global__ __launch_bounds__(256) void cvt_w_kernel(
    const float* __restrict__ W0, const float* __restrict__ W1,
    const float* __restrict__ W2, const float* __restrict__ W3,
    uint16_t* __restrict__ T0, uint16_t* __restrict__ T1,
    uint16_t* __restrict__ T2, uint16_t* __restrict__ T3) {
  const int p = blockIdx.y;
  const float* W = (p == 0) ? W0 : (p == 1) ? W1 : (p == 2) ? W2 : W3;
  uint16_t*    T = (p == 0) ? T0 : (p == 1) ? T1 : (p == 2) ? T2 : T3;
  int idx = blockIdx.x * 256 + threadIdx.x;
  int k = idx >> 9, n = idx & (EMB - 1);
  T[(size_t)n * EMB + k] = f2bf(W[idx]);
}

// ---------------- QKV projection ----------------
// Block = 128 rows x 64 cols; 8 waves split rows, SHARE a 64-col weight slab
// staged in LDS by async copies (double-buffered across the 3 projections).
__global__ __launch_bounds__(256) void qkv_kernel(
    const uint16_t* __restrict__ xb,
    const uint16_t* __restrict__ Wqt, const uint16_t* __restrict__ Wkt,
    const uint16_t* __restrict__ Wvt,
    const float* __restrict__ bq, const float* __restrict__ bk,
    const float* __restrict__ bv,
    uint16_t* __restrict__ qb, uint16_t* __restrict__ kb,
    uint16_t* __restrict__ vtb) {
  __shared__ uint16_t Wslab[2][64][EMB];       // 2 x 64KB
  const int t      = threadIdx.x;
  const int wave   = t >> 5;
  const int lane   = t & 31;
  const int lm     = lane & 15;
  const int laneHi = lane >> 4;
  const int row0   = blockIdx.x * 128 + wave * 16;
  const int c0     = blockIdx.y * 64;

  // prologue: async-stage Wq slab (64 contiguous rows of Wt = flat 64KB)
  {
    const uint16_t* src = Wqt + (size_t)c0 * EMB;
    const uint32_t dst = lds_off_of(&Wslab[0][0][0]);
#pragma unroll
    for (int i = 0; i < 16; ++i)
      async_load_b128(dst + t * 16 + i * 4096, src + t * 8 + i * 2048);
  }

  // cache A fragments (16 rows x all 16 K-windows), reused for q,k,v
  const uint16_t* xrow = xb + (size_t)(row0 + lm) * EMB;
  v16bf af[16];
#pragma unroll
  for (int kk = 0; kk < 16; ++kk) af[kk] = loadA(xrow + kk * 32, laneHi);

  const v8f z = {0, 0, 0, 0, 0, 0, 0, 0};
#pragma unroll
  for (int p = 0; p < 3; ++p) {
    wait_async0();        // our slab p writes have landed
    __syncthreads();      // everyone's landed; all waves done reading slab p-1
    if (p < 2) {          // prefetch slab p+1 into the other buffer
      const uint16_t* wsrc = ((p == 0) ? Wkt : Wvt) + (size_t)c0 * EMB;
      const uint32_t dst = lds_off_of(&Wslab[(p + 1) & 1][0][0]);
#pragma unroll
      for (int i = 0; i < 16; ++i)
        async_load_b128(dst + t * 16 + i * 4096, wsrc + t * 8 + i * 2048);
    }

    const uint16_t(*WS)[EMB] = Wslab[p & 1];
    const float* bias = (p == 0) ? bq : (p == 1) ? bk : bv;
    v8f acc[4] = {z, z, z, z};
#pragma unroll
    for (int kk = 0; kk < 16; ++kk) {
#pragma unroll
      for (int nt = 0; nt < 4; ++nt) {
        acc[nt] = wmma_bf16(af[kk],
                            loadB(&WS[nt * 16 + lm][kk * 32 + laneHi * 16]),
                            acc[nt]);
      }
    }
#pragma unroll
    for (int nt = 0; nt < 4; ++nt) {
      const int col  = c0 + nt * 16 + lm;
      const float bval = bias[col];
      const int h = col >> 6, d = col & 63;
#pragma unroll
      for (int r = 0; r < 8; ++r) {
        const int m  = row0 + r + laneHi * 8;
        const int bi = m >> 11;
        const int nn = m & (SEQ - 1);
        const uint16_t o = f2bf(acc[nt][r] + bval);
        if (p == 2) {
          vtb[((size_t)(bi * HEADS + h) * HD + d) * SEQ + nn] = o;   // [b][h][d][n]
        } else {
          uint16_t* dst = (p == 0) ? qb : kb;
          dst[((size_t)(bi * HEADS + h) * SEQ + nn) * HD + d] = o;   // [b][h][n][d]
        }
      }
    }
  }
}

// ---------------- flash attention ----------------
// Block = one (b,h) x 128 query rows (8 waves x 16 rows). K/V chunks (32 keys)
// are async-staged to LDS ONCE per block and shared by all 8 waves,
// double-buffered against compute.
__global__ __launch_bounds__(256) void attn_kernel(
    const uint16_t* __restrict__ qb, const uint16_t* __restrict__ kb,
    const uint16_t* __restrict__ vtb, uint16_t* __restrict__ ob) {
  __shared__ uint16_t Kbuf[2][32][HD];   // 2 x 4KB  (keys x d)
  __shared__ uint16_t Vbuf[2][HD][32];   // 2 x 4KB  (d x keys)
  __shared__ uint16_t Pst[8][16][32];    // per-wave P staging (C->A relayout)

  const int t      = threadIdx.x;
  const int wave   = t >> 5;
  const int lane   = t & 31;
  const int lm     = lane & 15;
  const int laneHi = lane >> 4;
  const int bh     = blockIdx.x >> 4;            // 0..31
  const int rt0    = (blockIdx.x & 15) * 128 + wave * 16;
  const int b      = bh >> 3, h = bh & 7;

  const uint16_t* kb_bh  = kb  + (size_t)bh * SEQ * HD;
  const uint16_t* vtb_bh = vtb + (size_t)bh * HD * SEQ;

  // chunk copy: K chunk is flat 4KB; V chunk is 64 rows x 64B from Vt
  auto issue_chunk = [&](int jc, int buf) {
    const uint32_t kdst = lds_off_of(&Kbuf[buf][0][0]);
    const uint32_t vdst = lds_off_of(&Vbuf[buf][0][0]);
    async_load_b128(kdst + t * 16, kb_bh + (size_t)jc * HD + t * 8);
    async_load_b128(vdst + t * 16,
                    vtb_bh + (size_t)(t >> 2) * SEQ + jc + (t & 3) * 8);
  };

  issue_chunk(0, 0);   // prologue

  const uint16_t* qrow = qb + ((size_t)bh * SEQ + rt0 + lm) * HD;
  const v16bf qf0 = loadA(qrow, laneHi);        // d = 0..31
  const v16bf qf1 = loadA(qrow + 32, laneHi);   // d = 32..63

  const v8f z = {0, 0, 0, 0, 0, 0, 0, 0};
  v8f acc[4] = {z, z, z, z};
  float mrow[8], lrow[8];
#pragma unroll
  for (int r = 0; r < 8; ++r) { mrow[r] = -3.0e38f; lrow[r] = 0.f; }

  uint16_t(*P)[32] = Pst[wave];

  for (int ic = 0; ic < SEQ / 32; ++ic) {
    const int cur = ic & 1;
    wait_async0();      // our async writes for chunk `cur` landed
    __syncthreads();    // all landed; all waves done reading buffer `1-cur`
    if (ic + 1 < SEQ / 32) issue_chunk((ic + 1) * 32, 1 - cur);

    // ---- S = Q Kt for 32 keys (two 16x16 tiles), K from LDS ----
    v8f s0 = z, s1 = z;
    s0 = wmma_bf16(qf0, loadB(&Kbuf[cur][lm][laneHi * 16]), s0);
    s0 = wmma_bf16(qf1, loadB(&Kbuf[cur][lm][32 + laneHi * 16]), s0);
    s1 = wmma_bf16(qf0, loadB(&Kbuf[cur][16 + lm][laneHi * 16]), s1);
    s1 = wmma_bf16(qf1, loadB(&Kbuf[cur][16 + lm][32 + laneHi * 16]), s1);

    // ---- online softmax (reference: softmax of UNSCALED logits) ----
    float p0v[8], p1v[8], alpha[8];
#pragma unroll
    for (int r = 0; r < 8; ++r) {
      float tmx = rmax16(fmaxf(s0[r], s1[r]));
      float mn  = fmaxf(mrow[r], tmx);
      float al  = __expf(mrow[r] - mn);
      float e0  = __expf(s0[r] - mn);
      float e1  = __expf(s1[r] - mn);
      float rs  = rsum16(e0 + e1);
      lrow[r]   = lrow[r] * al + rs;
      mrow[r]   = mn;
      alpha[r]  = al;
      p0v[r] = e0;
      p1v[r] = e1;
    }
#pragma unroll
    for (int nt = 0; nt < 4; ++nt)
#pragma unroll
      for (int r = 0; r < 8; ++r) acc[nt][r] *= alpha[r];

    // ---- stage P (C layout -> bf16 A layout, per-wave LDS tile) ----
#pragma unroll
    for (int r = 0; r < 8; ++r) {
      const int m = r + laneHi * 8;
      P[m][lm]      = f2bf(p0v[r]);
      P[m][16 + lm] = f2bf(p1v[r]);
    }
    asm volatile("" ::: "memory");   // same-wave LDS is in-order
    const v16bf pf = loadA(&P[lm][0], laneHi);
    asm volatile("" ::: "memory");

    // ---- O += P @ V, V from LDS ----
#pragma unroll
    for (int nt = 0; nt < 4; ++nt)
      acc[nt] = wmma_bf16(pf, loadB(&Vbuf[cur][nt * 16 + lm][laneHi * 16]),
                          acc[nt]);
  }

  // reference divides attention weights by sqrt(EMB) AFTER softmax
  const float inv_sqrt_e = 0.04419417382415922f;  // 1/sqrt(512)
#pragma unroll
  for (int nt = 0; nt < 4; ++nt) {
    const int col = h * HD + nt * 16 + lm;
#pragma unroll
    for (int r = 0; r < 8; ++r) {
      const int nn = rt0 + r + laneHi * 8;
      const float val = acc[nt][r] * (inv_sqrt_e / lrow[r]);
      ob[((size_t)b * SEQ + nn) * EMB + col] = f2bf(val);
    }
  }
}

// ---------------- output projection: out = O @ Wp + bp (fp32 out) ----------------
__global__ __launch_bounds__(256) void oproj_kernel(
    const uint16_t* __restrict__ ob, const uint16_t* __restrict__ Wpt,
    const float* __restrict__ bp, float* __restrict__ out) {
  __shared__ uint16_t Wslab[64][EMB];   // 64KB shared weight slab
  const int t      = threadIdx.x;
  const int wave   = t >> 5;
  const int lane   = t & 31;
  const int lm     = lane & 15;
  const int laneHi = lane >> 4;
  const int row0   = blockIdx.x * 128 + wave * 16;
  const int c0     = blockIdx.y * 64;

  {
    const uint16_t* src = Wpt + (size_t)c0 * EMB;
    const uint32_t dst = lds_off_of(&Wslab[0][0]);
#pragma unroll
    for (int i = 0; i < 16; ++i)
      async_load_b128(dst + t * 16 + i * 4096, src + t * 8 + i * 2048);
  }

  const uint16_t* orow = ob + (size_t)(row0 + lm) * EMB;
  v16bf af[16];
#pragma unroll
  for (int kk = 0; kk < 16; ++kk) af[kk] = loadA(orow + kk * 32, laneHi);

  wait_async0();
  __syncthreads();

  const v8f z = {0, 0, 0, 0, 0, 0, 0, 0};
  v8f acc[4] = {z, z, z, z};
#pragma unroll
  for (int kk = 0; kk < 16; ++kk) {
#pragma unroll
    for (int nt = 0; nt < 4; ++nt) {
      acc[nt] = wmma_bf16(af[kk],
                          loadB(&Wslab[nt * 16 + lm][kk * 32 + laneHi * 16]),
                          acc[nt]);
    }
  }
#pragma unroll
  for (int nt = 0; nt < 4; ++nt) {
    const int col  = c0 + nt * 16 + lm;
    const float bval = bp[col];
#pragma unroll
    for (int r = 0; r < 8; ++r) {
      const int m = row0 + r + laneHi * 8;
      out[(size_t)m * EMB + col] = acc[nt][r] + bval;
    }
  }
}

// ---------------- host launcher ----------------
extern "C" void kernel_launch(void* const* d_in, const int* in_sizes, int n_in,
                              void* d_out, int out_size, void* d_ws, size_t ws_size,
                              hipStream_t stream) {
  const float* x  = (const float*)d_in[0];
  const float* Wq = (const float*)d_in[1];
  const float* bq = (const float*)d_in[2];
  const float* Wk = (const float*)d_in[3];
  const float* bk = (const float*)d_in[4];
  const float* Wv = (const float*)d_in[5];
  const float* bv = (const float*)d_in[6];
  const float* Wp = (const float*)d_in[7];
  const float* bp = (const float*)d_in[8];
  float* out = (float*)d_out;

  uint8_t* ws = (uint8_t*)d_ws;
  size_t off = 0;
  const size_t actBytes = (size_t)ROWS * EMB * 2;   // 8 MB each
  const size_t wBytes   = (size_t)EMB * EMB * 2;    // 0.5 MB each
  uint16_t* xb   = (uint16_t*)(ws + off); off += actBytes;
  uint16_t* Wqt  = (uint16_t*)(ws + off); off += wBytes;
  uint16_t* Wkt  = (uint16_t*)(ws + off); off += wBytes;
  uint16_t* Wvt  = (uint16_t*)(ws + off); off += wBytes;
  uint16_t* Wpt  = (uint16_t*)(ws + off); off += wBytes;
  uint16_t* qb   = (uint16_t*)(ws + off); off += actBytes;
  uint16_t* kb   = (uint16_t*)(ws + off); off += actBytes;
  uint16_t* vtb  = (uint16_t*)(ws + off); off += actBytes;
  uint16_t* obuf = (uint16_t*)(ws + off); off += actBytes;
  (void)in_sizes; (void)n_in; (void)out_size; (void)ws_size;

  cvt_x_kernel<<<(ROWS * EMB / 4) / 256, 256, 0, stream>>>(x, xb);
  cvt_w_kernel<<<dim3(EMB * EMB / 256, 4), 256, 0, stream>>>(Wq, Wk, Wv, Wp,
                                                             Wqt, Wkt, Wvt, Wpt);
  qkv_kernel<<<dim3(ROWS / 128, EMB / 64), 256, 0, stream>>>(
      xb, Wqt, Wkt, Wvt, bq, bk, bv, qb, kb, vtb);
  attn_kernel<<<BATCH * HEADS * (SEQ / 128), 256, 0, stream>>>(qb, kb, vtb, obuf);
  oproj_kernel<<<dim3(ROWS / 128, EMB / 64), 256, 0, stream>>>(obuf, Wpt, bp, out);
}
